// SwitchGate_67130338837015
// MI455X (gfx1250) — matrix operand order repaired and checked by reading, not compiled, not run on hardware
//
#include <hip/hip_runtime.h>
#include <hip/hip_bf16.h>

typedef __bf16        v16bf __attribute__((ext_vector_type(16)));
typedef float         v8f   __attribute__((ext_vector_type(8)));
typedef float         v4f   __attribute__((ext_vector_type(4)));
typedef unsigned int  v8u   __attribute__((ext_vector_type(8)));

union BFrag { v8u u; v16bf bf; };

#define DIMK     2048
#define NEXP     64
#define NTOK     32768
#define KCHUNKS  64          // 2048 / 32
#define CAPACITY 32768.0f
#define EPS      1e-6f

// ---------------------------------------------------------------------------
// Split f32 pair -> packed bf16 hi word + bf16 lo (residual) word.
// Element 0 -> low 16 bits, element 1 -> high 16 bits (WMMA halfword order).
// Truncation is fine: the residual term recovers ~16 mantissa bits overall.
// ---------------------------------------------------------------------------
__device__ __forceinline__ void pack2(float e0, float e1, unsigned& h, unsigned& l) {
    unsigned b0 = __float_as_uint(e0), b1 = __float_as_uint(e1);
    h = (b0 >> 16) | (b1 & 0xffff0000u);
    float r0 = e0 - __uint_as_float(b0 & 0xffff0000u);
    float r1 = e1 - __uint_as_float(b1 & 0xffff0000u);
    l = (__float_as_uint(r0) >> 16) | (__float_as_uint(r1) & 0xffff0000u);
}

// ---------------------------------------------------------------------------
// Kernel 0: repack w_gate [2048,64] f32 into WMMA B-fragment order, bf16 hi/lo.
// B frag (16x16x32 bf16): lane L holds column n0 + L%16; halfword hw holds
// K = kchunk*32 + (L/16)*8 + (hw&7) + (hw>>3)*16.  Fragment (c,t) stored contiguously:
// dst = ((c*4 + t)*32 + L)*16 + hw   (32 bytes per lane -> one v8u load later)
// ---------------------------------------------------------------------------
__global__ void wprep_kernel(const float* __restrict__ w,
                             unsigned short* __restrict__ hi,
                             unsigned short* __restrict__ lo) {
    int gid = blockIdx.x * 256 + threadIdx.x;   // [0, 64*4*32*16) = 131072
    int hw  =  gid        & 15;
    int L   = (gid >> 4)  & 31;
    int t   = (gid >> 9)  & 3;
    int c   =  gid >> 11;
    int k   = c * 32 + ((L >> 4) << 3) + (hw & 7) + ((hw >> 3) << 4);
    int n   = t * 16 + (L & 15);
    float v = w[k * NEXP + n];
    unsigned vb = __float_as_uint(v);
    float r = v - __uint_as_float(vb & 0xffff0000u);
    hi[gid] = (unsigned short)(vb >> 16);
    lo[gid] = (unsigned short)(__float_as_uint(r) >> 16);
}

// ---------------------------------------------------------------------------
// Kernel 1: bf16x3 split-precision GEMM (logits) + per-row softmax reduction.
// Block = 256 threads = 8 waves. Wave owns 2 strips of 16 rows x all 64 cols.
// Emits per-token top1 index and score = 1/sum(exp(l - max)).
// ---------------------------------------------------------------------------
__global__ __launch_bounds__(256) void gate_gemm_kernel(
        const float* __restrict__ x, const float* __restrict__ bgate,
        const unsigned int* __restrict__ wBhi, const unsigned int* __restrict__ wBlo,
        int* __restrict__ topIdx, float* __restrict__ topVal) {
    const int lane = threadIdx.x & 31;
    const int wave = threadIdx.x >> 5;
    const int l16  = lane & 15;
    const int half = lane >> 4;                       // 0 or 1
    const int mBlock = blockIdx.x * 256 + wave * 32;  // 2 strips * 16 rows

    const v8u* Bhi = (const v8u*)wBhi;   // 32B fragments, index (c*4+t)*32+lane
    const v8u* Blo = (const v8u*)wBlo;

    v8f acc[2][4];
    #pragma unroll
    for (int s = 0; s < 2; ++s)
        #pragma unroll
        for (int t = 0; t < 4; ++t)
            acc[s][t] = (v8f){0.f,0.f,0.f,0.f,0.f,0.f,0.f,0.f};

    // A frag (16x32 bf16): lane holds row m0 + L%16; halfwords 0..7 cover
    // K = g0..g0+7 and halfwords 8..15 cover K = g0+16..g0+23, g0 = 8*(L/16).
    const float* xw = x + (size_t)(mBlock + l16) * DIMK + (half << 3);

    #pragma unroll 1
    for (int c = 0; c < KCHUNKS; ++c) {
        const int k0 = c * 32;
        BFrag bh[4], bl[4];
        #pragma unroll
        for (int t = 0; t < 4; ++t) {
            bh[t].u = Bhi[(c * 4 + t) * 32 + lane];
            bl[t].u = Blo[(c * 4 + t) * 32 + lane];
        }
        #pragma unroll
        for (int s = 0; s < 2; ++s) {
            const v4f* p = (const v4f*)(xw + (size_t)s * 16 * DIMK + k0);
            v4f A0 = __builtin_nontemporal_load(p);
            v4f A1 = __builtin_nontemporal_load(p + 1);
            v4f A2 = __builtin_nontemporal_load(p + 4);   // +16 floats
            v4f A3 = __builtin_nontemporal_load(p + 5);
            float fv[16];
            #pragma unroll
            for (int i = 0; i < 4; ++i) {
                fv[i] = A0[i]; fv[4 + i] = A1[i]; fv[8 + i] = A2[i]; fv[12 + i] = A3[i];
            }
            BFrag ah, al;
            #pragma unroll
            for (int i = 0; i < 8; ++i) {
                unsigned h, l;
                pack2(fv[2 * i], fv[2 * i + 1], h, l);
                ah.u[i] = h; al.u[i] = l;
            }
            #pragma unroll
            for (int t = 0; t < 4; ++t) {
                acc[s][t] = __builtin_amdgcn_wmma_f32_16x16x32_bf16(
                    false, ah.bf, false, bh[t].bf, (short)0, acc[s][t], false, false);
                acc[s][t] = __builtin_amdgcn_wmma_f32_16x16x32_bf16(
                    false, ah.bf, false, bl[t].bf, (short)0, acc[s][t], false, false);
                acc[s][t] = __builtin_amdgcn_wmma_f32_16x16x32_bf16(
                    false, al.bf, false, bh[t].bf, (short)0, acc[s][t], false, false);
            }
        }
    }

    float bias[4];
    #pragma unroll
    for (int t = 0; t < 4; ++t) bias[t] = bgate[t * 16 + l16];

    // C layout: lanes 0-15 hold N=lane, VGPR r -> row r; lanes 16-31 -> row 8+r.
    // Row m's 64 logits live in one 16-lane half x 4 accumulator tiles.
    #pragma unroll
    for (int s = 0; s < 2; ++s) {
        for (int r = 0; r < 8; ++r) {
            float bestV = acc[s][0][r] + bias[0];
            int   bestE = l16;
            #pragma unroll
            for (int t = 1; t < 4; ++t) {
                float v = acc[s][t][r] + bias[t];
                int   e = t * 16 + l16;
                if (v > bestV) { bestV = v; bestE = e; }
            }
            #pragma unroll
            for (int off = 1; off < 16; off <<= 1) {   // width-16 butterfly
                float ov = __shfl_xor(bestV, off, 16);
                int   oe = __shfl_xor(bestE, off, 16);
                if (ov > bestV || (ov == bestV && oe < bestE)) { bestV = ov; bestE = oe; }
            }
            float se = 0.f;
            #pragma unroll
            for (int t = 0; t < 4; ++t)
                se += __expf((acc[s][t][r] + bias[t]) - bestV);
            #pragma unroll
            for (int off = 1; off < 16; off <<= 1)
                se += __shfl_xor(se, off, 16);
            if (l16 == 0) {
                int m = mBlock + s * 16 + r + 8 * half;
                topIdx[m] = bestE;
                topVal[m] = 1.0f / se;     // softmax value at the argmax
            }
        }
    }
}

// ---------------------------------------------------------------------------
// Kernel 2: deterministic per-expert column sum (fixed-order tree reduction).
// ---------------------------------------------------------------------------
__global__ __launch_bounds__(256) void denom_kernel(
        const int* __restrict__ idx, const float* __restrict__ sval,
        float* __restrict__ denom) {
    __shared__ float red[256];
    int e = blockIdx.x;
    float p = 0.f;
    for (int n = threadIdx.x; n < NTOK; n += 256)
        p += (idx[n] == e) ? sval[n] : 0.f;
    red[threadIdx.x] = p;
    __syncthreads();
    for (int s = 128; s > 0; s >>= 1) {
        if ((int)threadIdx.x < s) red[threadIdx.x] += red[threadIdx.x + s];
        __syncthreads();
    }
    if (threadIdx.x == 0) denom[e] = red[0];
}

// ---------------------------------------------------------------------------
// Kernel 3: dense scatter of the normalized output, B128 nontemporal stores.
// ---------------------------------------------------------------------------
__global__ __launch_bounds__(256) void scatter_kernel(
        const int* __restrict__ idx, const float* __restrict__ sval,
        const float* __restrict__ denom, float* __restrict__ out) {
    __shared__ float dsh[NEXP];
    if (threadIdx.x < NEXP) dsh[threadIdx.x] = denom[threadIdx.x] + EPS;
    __syncthreads();
    int n  = blockIdx.x * 16 + (threadIdx.x >> 4);
    int e0 = (threadIdx.x & 15) * 4;
    int bi = idx[n];
    float scaled = sval[n] * CAPACITY;
    v4f o;
    #pragma unroll
    for (int j = 0; j < 4; ++j)
        o[j] = (e0 + j == bi) ? (scaled / dsh[e0 + j]) : 0.f;
    __builtin_nontemporal_store(o, (v4f*)(out + (size_t)n * NEXP + e0));
}

// ---------------------------------------------------------------------------
extern "C" void kernel_launch(void* const* d_in, const int* in_sizes, int n_in,
                              void* d_out, int out_size, void* d_ws, size_t ws_size,
                              hipStream_t stream) {
    const float* x = (const float*)d_in[0];   // [32768, 2048]
    const float* w = (const float*)d_in[1];   // [2048, 64]
    const float* b = (const float*)d_in[2];   // [64]
    float* out = (float*)d_out;               // [32768, 64]

    char* ws = (char*)d_ws;
    unsigned short* wBhi = (unsigned short*)ws;              // 256 KB
    unsigned short* wBlo = (unsigned short*)(ws + 262144);   // 256 KB
    int*   topIdx = (int*)  (ws + 524288);                   // 128 KB
    float* topVal = (float*)(ws + 655360);                   // 128 KB
    float* denom  = (float*)(ws + 786432);                   // 256 B

    wprep_kernel<<<512, 256, 0, stream>>>(w, wBhi, wBlo);
    gate_gemm_kernel<<<128, 256, 0, stream>>>(x, b,
        (const unsigned int*)wBhi, (const unsigned int*)wBlo, topIdx, topVal);
    denom_kernel<<<NEXP, 256, 0, stream>>>(topIdx, topVal, denom);
    scatter_kernel<<<NTOK / 16, 256, 0, stream>>>(topIdx, topVal, denom, out);
}
